// BagModel_40630390620760
// MI455X (gfx1250) — matrix operand order repaired and checked by reading, not compiled, not run on hardware
//
#include <hip/hip_runtime.h>
#include <hip/hip_bf16.h>

// ---------------------------------------------------------------------------
// BagModel on MI455X (gfx1250): fused GEMM1+ReLU+segment-sum, then GEMM2,
// both on V_WMMA_F32_16X16X32_BF16 with split-bf16 (hi+lo) for ~fp32 accuracy.
// Round 2: run-length-compressed segment atomics (ids are sorted) and b128
// vectorized loads/stores on the dominant x stream.
// ---------------------------------------------------------------------------

typedef __attribute__((ext_vector_type(16))) __bf16          v16bf;
typedef __attribute__((ext_vector_type(8)))  float           v8f;
typedef __attribute__((ext_vector_type(8)))  unsigned short  ushort8;
typedef __attribute__((ext_vector_type(4)))  unsigned short  ushort4v;

union Frag16 { v16bf bf; ushort8 h[2]; };

__device__ __forceinline__ unsigned short f32_to_bf16_rne(float f) {
    unsigned u = __float_as_uint(f);
    unsigned r = 0x7FFFu + ((u >> 16) & 1u);
    return (unsigned short)((u + r) >> 16);
}
__device__ __forceinline__ float bf16_to_f32(unsigned short h) {
    return __uint_as_float(((unsigned)h) << 16);
}

// Problem constants (from reference)
#define NROWS 500000
#define DDIM  128
#define HDIM  256
#define ODIM  64
#define NBAGS 10000

// LDS pitches (in ushorts) padded to break 64-bank stride conflicts,
// kept as multiples of 8 halves (16B) so ds_load_b128 stays aligned.
#define XPITCH 136   // 128 + 8   -> 272B row stride
#define MPITCH 264   // 256 + 8   -> 528B row stride

// ---------------------------------------------------------------------------
__global__ void k_zero(float* __restrict__ p, int n) {
    int i = blockIdx.x * blockDim.x + threadIdx.x;
    if (i < n) p[i] = 0.0f;
}

// W1 [128,256] f32 -> W1^T hi/lo bf16 [256,128]
__global__ void k_prep_w1(const float* __restrict__ W1,
                          unsigned short* __restrict__ hi,
                          unsigned short* __restrict__ lo) {
    int i = blockIdx.x * 256 + threadIdx.x;          // 0..32767
    int k = i >> 8;                                   // 0..127
    int c = i & 255;                                  // 0..255
    float v = W1[i];
    unsigned short h = f32_to_bf16_rne(v);
    hi[c * DDIM + k] = h;
    lo[c * DDIM + k] = f32_to_bf16_rne(v - bf16_to_f32(h));
}

// W2 [256,64] f32 -> W2^T hi/lo bf16 [64,256]
__global__ void k_prep_w2(const float* __restrict__ W2,
                          unsigned short* __restrict__ hi,
                          unsigned short* __restrict__ lo) {
    int i = blockIdx.x * 256 + threadIdx.x;          // 0..16383
    int k = i >> 6;                                   // 0..255
    int c = i & 63;                                   // 0..63
    float v = W2[i];
    unsigned short h = f32_to_bf16_rne(v);
    hi[c * HDIM + k] = h;
    lo[c * HDIM + k] = f32_to_bf16_rne(v - bf16_to_f32(h));
}

// ---------------------------------------------------------------------------
// Fused: 16-row tile of h = relu(x@W1+b1), segment-added into sums.
// Block = 256 threads = 8 waves; each wave owns two 16-column N-tiles
// (8*2*16 = 256 = H). K=128 -> 4 WMMA K-steps of 32; 3 WMMAs per step.
// ids are sorted, so consecutive rows of a tile usually share a bag:
// run-length-compress the per-column partial sums before hitting L2 atomics.
// ---------------------------------------------------------------------------
__global__ __launch_bounds__(256, 2)
void k_gemm1(const float* __restrict__ x, const int* __restrict__ ids,
             const float* __restrict__ b1,
             const unsigned short* __restrict__ w1th,
             const unsigned short* __restrict__ w1tl,
             float* __restrict__ sums, float* __restrict__ counts)
{
    __shared__ __align__(16) unsigned short xh[16 * XPITCH];
    __shared__ __align__(16) unsigned short xl[16 * XPITCH];
    __shared__ int bag[16];

    const int tid = threadIdx.x;
    const int m0  = blockIdx.x * 16;

    // Load 16x128 f32 tile of x with b128 loads; split into bf16 hi/lo in LDS.
    #pragma unroll
    for (int i = 0; i < 2; ++i) {
        const int e   = (tid + i * 256) * 4;          // 0..2044, step 4
        const int row = e >> 7;
        const int k   = e & 127;
        const float4 v = *(const float4*)&x[(size_t)(m0 + row) * DDIM + k];
        ushort4v h, l;
        h.x = f32_to_bf16_rne(v.x); l.x = f32_to_bf16_rne(v.x - bf16_to_f32(h.x));
        h.y = f32_to_bf16_rne(v.y); l.y = f32_to_bf16_rne(v.y - bf16_to_f32(h.y));
        h.z = f32_to_bf16_rne(v.z); l.z = f32_to_bf16_rne(v.z - bf16_to_f32(h.z));
        h.w = f32_to_bf16_rne(v.w); l.w = f32_to_bf16_rne(v.w - bf16_to_f32(h.w));
        *(ushort4v*)&xh[row * XPITCH + k] = h;
        *(ushort4v*)&xl[row * XPITCH + k] = l;
    }
    if (tid < 16) bag[tid] = ids[m0 + tid];
    __syncthreads();

    const int  wave   = tid >> 5;
    const int  lane   = tid & 31;
    const int  ln     = lane & 15;
    const bool hiHalf = lane >= 16;
    const int  akb    = hiHalf ? 8  : 0;   // A-fragment K base (16-bit A layout)
    const int  bkb    = hiHalf ? 8  : 0;   // B-fragment K base

    #pragma unroll
    for (int t = 0; t < 2; ++t) {
        const int nt  = wave * 2 + t;            // N-tile 0..15
        const int col = nt * 16 + ln;            // global H column for this lane
        v8f c = {};
        #pragma unroll
        for (int s = 0; s < 4; ++s) {
            const int ka = s * 32 + akb;
            Frag16 ah, al, bh, bl;
            ah.h[0] = *(const ushort8*)&xh[ln * XPITCH + ka];
            ah.h[1] = *(const ushort8*)&xh[ln * XPITCH + ka + 16];
            al.h[0] = *(const ushort8*)&xl[ln * XPITCH + ka];
            al.h[1] = *(const ushort8*)&xl[ln * XPITCH + ka + 16];
            const int kb = s * 32 + bkb;
            bh.h[0] = *(const ushort8*)&w1th[(size_t)col * DDIM + kb];
            bh.h[1] = *(const ushort8*)&w1th[(size_t)col * DDIM + kb + 16];
            bl.h[0] = *(const ushort8*)&w1tl[(size_t)col * DDIM + kb];
            bl.h[1] = *(const ushort8*)&w1tl[(size_t)col * DDIM + kb + 16];
            c = __builtin_amdgcn_wmma_f32_16x16x32_bf16(false, ah.bf, false, bh.bf, (short)0, c, false, false);
            c = __builtin_amdgcn_wmma_f32_16x16x32_bf16(false, ah.bf, false, bl.bf, (short)0, c, false, false);
            c = __builtin_amdgcn_wmma_f32_16x16x32_bf16(false, al.bf, false, bh.bf, (short)0, c, false, false);
        }
        const float bias = b1[col];
        // Run-length compressed segment accumulation: rows m..m+7 of this
        // half live in consecutive elements of c.  bag[] is sorted, so flush
        // to L2 atomics only at segment boundaries (predicate is uniform
        // across each 16-lane half-wave).
        float acc = 0.0f;
        #pragma unroll
        for (int r = 0; r < 8; ++r) {
            const int m = r + (hiHalf ? 8 : 0);  // C layout: VGPR r, lane half
            float v = c[r] + bias;
            acc += (v > 0.0f ? v : 0.0f);
            const bool flush = (r == 7) || (bag[m + 1] != bag[m]);
            if (flush) {
                __hip_atomic_fetch_add(&sums[(size_t)bag[m] * HDIM + col], acc,
                                       __ATOMIC_RELAXED, __HIP_MEMORY_SCOPE_AGENT);
                acc = 0.0f;
            }
        }
    }
    if (tid < 16)
        __hip_atomic_fetch_add(&counts[bag[tid]], 1.0f,
                               __ATOMIC_RELAXED, __HIP_MEMORY_SCOPE_AGENT);
}

// ---------------------------------------------------------------------------
// GEMM2: means[16 bags,256] @ W2[256,64] + b2.  Block = 128 threads = 4 waves,
// one 16-col N-tile per wave.  K=256 -> 8 WMMA K-steps, 3 WMMAs per step.
// ---------------------------------------------------------------------------
__global__ __launch_bounds__(128, 4)
void k_gemm2(const float* __restrict__ sums, const float* __restrict__ counts,
             const float* __restrict__ b2,
             const unsigned short* __restrict__ w2th,
             const unsigned short* __restrict__ w2tl,
             float* __restrict__ out)
{
    __shared__ __align__(16) unsigned short mh[16 * MPITCH];
    __shared__ __align__(16) unsigned short ml[16 * MPITCH];
    __shared__ float inv[16];

    const int tid = threadIdx.x;
    const int m0  = blockIdx.x * 16;

    if (tid < 16) inv[tid] = 1.0f / fmaxf(counts[m0 + tid], 1.0f);
    __syncthreads();

    #pragma unroll
    for (int i = 0; i < 8; ++i) {
        const int e   = (tid + i * 128) * 4;          // 0..4092, step 4
        const int row = e >> 8;
        const int k   = e & 255;
        float4 v = *(const float4*)&sums[(size_t)(m0 + row) * HDIM + k];
        const float sc = inv[row];
        v.x *= sc; v.y *= sc; v.z *= sc; v.w *= sc;
        ushort4v h, l;
        h.x = f32_to_bf16_rne(v.x); l.x = f32_to_bf16_rne(v.x - bf16_to_f32(h.x));
        h.y = f32_to_bf16_rne(v.y); l.y = f32_to_bf16_rne(v.y - bf16_to_f32(h.y));
        h.z = f32_to_bf16_rne(v.z); l.z = f32_to_bf16_rne(v.z - bf16_to_f32(h.z));
        h.w = f32_to_bf16_rne(v.w); l.w = f32_to_bf16_rne(v.w - bf16_to_f32(h.w));
        *(ushort4v*)&mh[row * MPITCH + k] = h;
        *(ushort4v*)&ml[row * MPITCH + k] = l;
    }
    __syncthreads();

    const int  wave   = tid >> 5;            // N-tile 0..3
    const int  lane   = tid & 31;
    const int  ln     = lane & 15;
    const bool hiHalf = lane >= 16;
    const int  akb    = hiHalf ? 8  : 0;
    const int  bkb    = hiHalf ? 8  : 0;
    const int  col    = wave * 16 + ln;      // output column 0..63

    v8f c = {};
    #pragma unroll
    for (int s = 0; s < 8; ++s) {
        const int ka = s * 32 + akb;
        Frag16 ah, al, bh, bl;
        ah.h[0] = *(const ushort8*)&mh[ln * MPITCH + ka];
        ah.h[1] = *(const ushort8*)&mh[ln * MPITCH + ka + 16];
        al.h[0] = *(const ushort8*)&ml[ln * MPITCH + ka];
        al.h[1] = *(const ushort8*)&ml[ln * MPITCH + ka + 16];
        const int kb = s * 32 + bkb;
        bh.h[0] = *(const ushort8*)&w2th[(size_t)col * HDIM + kb];
        bh.h[1] = *(const ushort8*)&w2th[(size_t)col * HDIM + kb + 16];
        bl.h[0] = *(const ushort8*)&w2tl[(size_t)col * HDIM + kb];
        bl.h[1] = *(const ushort8*)&w2tl[(size_t)col * HDIM + kb + 16];
        c = __builtin_amdgcn_wmma_f32_16x16x32_bf16(false, ah.bf, false, bh.bf, (short)0, c, false, false);
        c = __builtin_amdgcn_wmma_f32_16x16x32_bf16(false, ah.bf, false, bl.bf, (short)0, c, false, false);
        c = __builtin_amdgcn_wmma_f32_16x16x32_bf16(false, al.bf, false, bh.bf, (short)0, c, false, false);
    }
    const float bias = b2[col];
    #pragma unroll
    for (int r = 0; r < 8; ++r) {
        const int m = r + (hiHalf ? 8 : 0);
        out[(size_t)(m0 + m) * ODIM + col] = c[r] + bias;
    }
}

// ---------------------------------------------------------------------------
extern "C" void kernel_launch(void* const* d_in, const int* in_sizes, int n_in,
                              void* d_out, int out_size, void* d_ws, size_t ws_size,
                              hipStream_t stream)
{
    (void)in_sizes; (void)n_in; (void)out_size; (void)ws_size;

    const float* x   = (const float*)d_in[0];
    const int*   ids = (const int*)  d_in[1];
    const float* W1  = (const float*)d_in[2];
    const float* b1  = (const float*)d_in[3];
    const float* W2  = (const float*)d_in[4];
    const float* b2  = (const float*)d_in[5];
    float* out = (float*)d_out;

    // Workspace layout (bytes):
    //   [0      ,  64K)  W1^T hi bf16 [256][128]
    //   [64K    , 128K)  W1^T lo
    //   [128K   , 160K)  W2^T hi bf16 [64][256]
    //   [160K   , 192K)  W2^T lo
    //   [192K   , +10.24M)  sums f32 [10000][256]
    //   then 40K           counts f32 [10000]
    char* ws = (char*)d_ws;
    unsigned short* w1th = (unsigned short*)(ws);
    unsigned short* w1tl = (unsigned short*)(ws + (64u << 10));
    unsigned short* w2th = (unsigned short*)(ws + (128u << 10));
    unsigned short* w2tl = (unsigned short*)(ws + (160u << 10));
    float* sums   = (float*)(ws + (192u << 10));
    float* counts = (float*)(ws + (192u << 10) + (size_t)NBAGS * HDIM * sizeof(float));

    const int nzero = NBAGS * HDIM + NBAGS;          // sums+counts contiguous
    k_zero<<<(nzero + 255) / 256, 256, 0, stream>>>(sums, nzero);
    k_prep_w1<<<(DDIM * HDIM) / 256, 256, 0, stream>>>(W1, w1th, w1tl);
    k_prep_w2<<<(HDIM * ODIM) / 256, 256, 0, stream>>>(W2, w2th, w2tl);
    k_gemm1<<<NROWS / 16, 256, 0, stream>>>(x, ids, b1, w1th, w1tl, sums, counts);
    k_gemm2<<<NBAGS / 16, 128, 0, stream>>>(sums, counts, b2, w2th, w2tl, out);
}
